// attention5_7009386627376
// MI455X (gfx1250) — matrix operand order repaired and checked by reading, not compiled, not run on hardware
//
#include <hip/hip_runtime.h>
#include <hip/hip_bf16.h>

typedef __bf16 v16bf __attribute__((ext_vector_type(16)));
typedef __bf16 v8bf  __attribute__((ext_vector_type(8)));
typedef float  v8f   __attribute__((ext_vector_type(8)));
typedef unsigned int u32x4 __attribute__((ext_vector_type(4)));
typedef int          i32x8 __attribute__((ext_vector_type(8)));
typedef int          i32x4 __attribute__((ext_vector_type(4)));
typedef __attribute__((address_space(3))) unsigned char lds_char;

#define NROW 4096
#define DDIM 256
#define LEAKY 0.2f

#if __has_builtin(__builtin_amdgcn_tensor_load_to_lds) && \
    __has_builtin(__builtin_amdgcn_s_wait_tensorcnt)
#define HAVE_TDM 1
#else
#define HAVE_TDM 0
#endif

// A-panel staging in LDS: 16 rows x 128 bf16, padded 256B->272B per row so the
// 16 fragment rows start on distinct bank groups (row -> dword offset row*68).
#define PANEL_ROW_B   272
#define PANEL_BYTES   (16 * PANEL_ROW_B)

// ---------------------------------------------------------------------------
// Tiled fp32 transpose: out[c][r] = in[r][c].  grid = (C/32, R/32), block (32,8)
// ---------------------------------------------------------------------------
__global__ void transpose_f32(const float* __restrict__ in, float* __restrict__ out,
                              int R, int C) {
  __shared__ float t[32][33];
  int tx = threadIdx.x, ty = threadIdx.y;
  int x  = blockIdx.x * 32 + tx;
  int y0 = blockIdx.y * 32;
  for (int i = 0; i < 32; i += 8)
    t[ty + i][tx] = in[(size_t)(y0 + ty + i) * C + x];
  __syncthreads();
  int ox  = y0 + tx;
  int oy0 = blockIdx.x * 32;
  for (int i = 0; i < 32; i += 8)
    out[(size_t)(oy0 + ty + i) * R + ox] = t[tx][ty + i];
}

__global__ void transpose_to_bf16(const float* __restrict__ in, __bf16* __restrict__ out,
                                  int R, int C) {
  __shared__ float t[32][33];
  int tx = threadIdx.x, ty = threadIdx.y;
  int x  = blockIdx.x * 32 + tx;
  int y0 = blockIdx.y * 32;
  for (int i = 0; i < 32; i += 8)
    t[ty + i][tx] = in[(size_t)(y0 + ty + i) * C + x];
  __syncthreads();
  int ox  = y0 + tx;
  int oy0 = blockIdx.x * 32;
  for (int i = 0; i < 32; i += 8)
    out[(size_t)(oy0 + ty + i) * R + ox] = (__bf16)t[tx][ty + i];
}

// ---------------------------------------------------------------------------
// Row-wise sparsemax of (-in[row,:]) over 4096 cols, tau via bisection.
// Register-resident: each thread keeps its 16 elements in VGPRs; reductions
// are wave32 shfl_xor + an 8-float LDS hop (no big LDS array, few barriers).
// fp32 result may alias `in` in place; bf16 copy for the WMMA GEMM.
// ---------------------------------------------------------------------------
__global__ __launch_bounds__(256) void sparsemax_rows(
    const float* __restrict__ in, float* __restrict__ outf, __bf16* __restrict__ outb) {
  __shared__ float red[8];
  const int tid  = threadIdx.x;
  const int wid  = tid >> 5, lane = tid & 31;
  const size_t row = blockIdx.x;
  const float* rp = in + row * NROW;

  float x[16];
  float lmax = -3.402823466e38f;
#pragma unroll
  for (int i = 0; i < 16; ++i) {
    x[i] = -rp[tid + (i << 8)];
    lmax = fmaxf(lmax, x[i]);
  }
  for (int off = 16; off; off >>= 1) lmax = fmaxf(lmax, __shfl_xor(lmax, off, 32));
  if (lane == 0) red[wid] = lmax;
  __syncthreads();
  float m = red[0];
#pragma unroll
  for (int w = 1; w < 8; ++w) m = fmaxf(m, red[w]);

  // bisection: f(tau) = sum max(x - tau, 0); f(m-1) >= 1, f(m) = 0
  float lo = m - 1.0f, hi = m;
  for (int it = 0; it < 32; ++it) {
    float tau = 0.5f * (lo + hi);
    float s = 0.0f;
#pragma unroll
    for (int i = 0; i < 16; ++i) s += fmaxf(x[i] - tau, 0.0f);
    for (int off = 16; off; off >>= 1) s += __shfl_xor(s, off, 32);
    __syncthreads();                 // red reusable
    if (lane == 0) red[wid] = s;
    __syncthreads();
    float tot = red[0];
#pragma unroll
    for (int w = 1; w < 8; ++w) tot += red[w];
    if (tot >= 1.0f) lo = tau; else hi = tau;   // uniform across block
  }
  const float tau = 0.5f * (lo + hi);

  float*  op = outf + row * NROW;
  __bf16* ob = outb + row * NROW;
#pragma unroll
  for (int i = 0; i < 16; ++i) {
    float v = fmaxf(x[i] - tau, 0.0f);
    op[tid + (i << 8)] = v;
    ob[tid + (i << 8)] = (__bf16)v;
  }
}

// ---------------------------------------------------------------------------
// TDM descriptor issue: 2D tensor load Global->LDS of a 16-row x 128-elem bf16
// panel (row stride K elems in memory), LDS-padded 16B per 256B row.
// D# layout per CDNA5 ISA ch.8: group0 = {count, lds_addr, global_addr, type=2},
// group1 = {data_size=2B, pad(interval=64dw, amount=4dw), tensor_dim0=K,
//           tensor_dim1=16, tile_dim0=128, tile_dim1=16, dim0_stride=K}.
// This toolchain exposes the 6-arg builtin:
//   (u32x4 g0, i32x8 g1, i32x4 g2, i32x4 g3, i32x8 g4, i32 cpol)
// ---------------------------------------------------------------------------
#if HAVE_TDM
__device__ inline void tdm_load_panel(unsigned long long gaddr, unsigned int ldsaddr,
                                      int K) {
  u32x4 g0;
  g0[0] = 1u;                                            // count=1 valid descriptor
  g0[1] = ldsaddr;                                       // lds_addr (bytes)
  g0[2] = (unsigned int)gaddr;                           // global_addr[31:0]
  g0[3] = ((unsigned int)(gaddr >> 32) & 0x01FFFFFFu)    // global_addr[56:32]
        | (2u << 30);                                    // type=2 ("image")
  i32x8 g1;
  g1[0] = (int)(0x00010000u                              // data_size=1 -> 2 bytes
        | (1u << 20)                                     // pad_enable
        | (5u << 22)                                     // pad_interval: 64 dwords
        | (3u << 25));                                   // pad_amount: 4 dwords
  g1[1] = (int)(((unsigned)K & 0xFFFFu) << 16);          // tensor_dim0[15:0]
  g1[2] = (int)((((unsigned)K >> 16) & 0xFFFFu)          // tensor_dim0[31:16]
        | (16u << 16));                                  // tensor_dim1 = 16
  g1[3] = (int)(128u << 16);                             // tile_dim0 = 128
  g1[4] = 16;                                            // tile_dim1 = 16
  g1[5] = K;                                             // tensor_dim0_stride lo32
  g1[6] = 0;
  g1[7] = 0;
  i32x4 gz4 = {0, 0, 0, 0};                              // groups 2/3 unused (2D)
  i32x8 gz8 = {0, 0, 0, 0, 0, 0, 0, 0};
  __builtin_amdgcn_tensor_load_to_lds(g0, g1, gz4, gz4, gz8, 0);
}
#endif

// ---------------------------------------------------------------------------
// WMMA bf16 GEMM with TDM A-panel staging: C[M,N] = A[M,K] * B[K,N] (BT[N,K]).
// grid.x = M/16; 512 threads = 16 waves = the 16 N-tiles of one M row-panel.
// A panel (16 x 128 bf16) is DMA'd into LDS by the Tensor Data Mover,
// double-buffered (issue chunk k+1, compute chunk k), handed off with
// s_wait_tensorcnt + workgroup barrier. B streams from L2 per wave.
// ---------------------------------------------------------------------------
__global__ __launch_bounds__(512) void gemm_bf16_tdm(
    const __bf16* __restrict__ A, const __bf16* __restrict__ BT,
    __bf16* __restrict__ C, int N, int K) {
  __shared__ __align__(16) unsigned char apanel[2][PANEL_BYTES];
  const int lane = threadIdx.x & 31;
  const int wave = threadIdx.x >> 5;
  const int m0 = blockIdx.x << 4;
  const int n0 = wave << 4;
  const int half = lane >> 4;
  const size_t brow = (size_t)(n0 + (lane & 15)) * K + (half << 4);
  const int nch = K >> 7;                       // 128-K chunks

  v8f c = {};

#if HAVE_TDM
  const unsigned long long abase =
      (unsigned long long)(size_t)(const void*)(A + (size_t)m0 * K);
  unsigned int loff[2];
  loff[0] = (unsigned int)(unsigned long long)(lds_char*)&apanel[0][0];
  loff[1] = (unsigned int)(unsigned long long)(lds_char*)&apanel[1][0];

  if (wave == 0) tdm_load_panel(abase, loff[0], K);
  for (int ch = 0; ch < nch; ++ch) {
    if (wave == 0) __builtin_amdgcn_s_wait_tensorcnt(0);
    __syncthreads();                            // buf[ch&1] filled & prior reads done
    if (wave == 0 && ch + 1 < nch)
      tdm_load_panel(abase + (((unsigned long long)(ch + 1)) << 8) /* *128*2B */,
                     loff[(ch + 1) & 1], K);    // overlap DMA with compute
    const unsigned char* arow =
        &apanel[ch & 1][0] + (lane & 15) * PANEL_ROW_B + (half << 5);
#pragma unroll
    for (int kc = 0; kc < 4; ++kc) {
      v16bf a, b;
      *(v8bf*)&a       = *(const v8bf*)(arow + kc * 64);        // K 16h+0..7
      *((v8bf*)&a + 1) = *(const v8bf*)(arow + kc * 64 + 16);   // K 16h+8..15
      b = *(const v16bf*)(BT + brow + (ch << 7) + (kc << 5));
      c = __builtin_amdgcn_wmma_f32_16x16x32_bf16(false, a, false, b,
                                                  (short)0, c, false, false);
    }
  }
#else
  // Fallback: cooperative synchronous staging into the same padded layout.
  const unsigned int* As = (const unsigned int*)A;
  unsigned int* pan32 = (unsigned int*)&apanel[0][0];
  for (int ch = 0; ch < nch; ++ch) {
    __syncthreads();
    for (int idx = threadIdx.x; idx < 1024; idx += 512) {
      int r = idx >> 6, d = idx & 63;
      pan32[r * 68 + d] = As[(size_t)(m0 + r) * (K >> 1) + (ch << 6) + d];
    }
    __syncthreads();
    const unsigned char* arow =
        &apanel[0][0] + (lane & 15) * PANEL_ROW_B + (half << 5);
#pragma unroll
    for (int kc = 0; kc < 4; ++kc) {
      v16bf a, b;
      *(v8bf*)&a       = *(const v8bf*)(arow + kc * 64);
      *((v8bf*)&a + 1) = *(const v8bf*)(arow + kc * 64 + 16);
      b = *(const v16bf*)(BT + brow + (ch << 7) + (kc << 5));
      c = __builtin_amdgcn_wmma_f32_16x16x32_bf16(false, a, false, b,
                                                  (short)0, c, false, false);
    }
  }
#endif

  // C layout: VGPR r -> M = m0 + r + 8*half, N = n0 + (lane&15)
  const int nc = n0 + (lane & 15);
  const int mb = m0 + (half << 3);
#pragma unroll
  for (int r = 0; r < 8; ++r)
    C[(size_t)(mb + r) * N + nc] = (__bf16)c[r];
}

// ---------------------------------------------------------------------------
// WMMA GEMM fused with G-epilogue: leakyrelu(X @ W + b), column partial sums
// accumulated into acc[256] via f32 atomics.  X: [4096,256] bf16,
// WT: [256,256] bf16 (W transposed), bias f32[256].  K=256 -> direct loads.
// ---------------------------------------------------------------------------
__global__ __launch_bounds__(256) void gemm_G_colsum(
    const __bf16* __restrict__ X, const __bf16* __restrict__ WT,
    const float* __restrict__ bias, float* __restrict__ acc) {
  const int K = DDIM, N = DDIM;
  const int lane = threadIdx.x & 31;
  const int wave = threadIdx.x >> 5;
  const int tile = blockIdx.x * 8 + wave;
  const int mt = tile / (N >> 4), nt = tile % (N >> 4);
  const int m0 = mt << 4, n0 = nt << 4;

  const int half = lane >> 4;
  const size_t arow = (size_t)(m0 + (lane & 15)) * K + (half << 4);
  const size_t brow = (size_t)(n0 + (lane & 15)) * K + (half << 4);

  v8f c = {};
  for (int k = 0; k < K; k += 32) {
    v16bf a, b;
    *(v8bf*)&a       = *(const v8bf*)(X + arow + k);
    *((v8bf*)&a + 1) = *(const v8bf*)(X + arow + k + 8);
    b = *(const v16bf*)(WT + brow + k);
    c = __builtin_amdgcn_wmma_f32_16x16x32_bf16(false, a, false, b,
                                                (short)0, c, false, false);
  }
  const int n = n0 + (lane & 15);
  const float bv = bias[n];
  float s = 0.0f;
#pragma unroll
  for (int r = 0; r < 8; ++r) {
    float v = c[r] + bv;
    v = (v > 0.0f) ? v : LEAKY * v;
    s += v;                                 // sum over this half's 8 rows
  }
  s += __shfl_xor(s, 16, 32);               // combine both M halves (wave32)
  if (lane < 16) atomicAdd(&acc[n], s);
}

// ---------------------------------------------------------------------------
// Small helpers
// ---------------------------------------------------------------------------
__global__ void zero_f32(float* p, int n) {
  int i = blockIdx.x * blockDim.x + threadIdx.x;
  if (i < n) p[i] = 0.0f;
}

__global__ void cosine_k(const float* __restrict__ acc, float* __restrict__ y) {
  __shared__ float r1[256], r2[256], r3[256];
  int tid = threadIdx.x;
  const float inv = 1.0f / 4096.0f;
  float v1 = acc[tid] * inv;
  float v2 = acc[256 + tid] * inv;
  r1[tid] = v1 * v2; r2[tid] = v1 * v1; r3[tid] = v2 * v2;
  __syncthreads();
  for (int s = 128; s > 0; s >>= 1) {
    if (tid < s) { r1[tid] += r1[tid + s]; r2[tid] += r2[tid + s]; r3[tid] += r3[tid + s]; }
    __syncthreads();
  }
  if (tid == 0) {
    float den = sqrtf(r2[0]) * sqrtf(r3[0]);
    y[0] = 1.0f - r1[0] / (den + 1e-8f);
  }
}

__global__ void fill_y(float4* __restrict__ out, const float* __restrict__ y, int n4) {
  float v = *y;
  float4 f = make_float4(v, v, v, v);
  for (int i = blockIdx.x * blockDim.x + threadIdx.x; i < n4; i += gridDim.x * blockDim.x)
    out[i] = f;
}

// ---------------------------------------------------------------------------
// Host orchestration
// ---------------------------------------------------------------------------
extern "C" void kernel_launch(void* const* d_in, const int* in_sizes, int n_in,
                              void* d_out, int out_size, void* d_ws, size_t ws_size,
                              hipStream_t stream) {
  (void)in_sizes; (void)n_in; (void)out_size; (void)ws_size;
  const float* row_vecs = (const float*)d_in[0];   // [4096,256]
  const float* col_vecs = (const float*)d_in[1];   // [4096,256]
  const float* cost     = (const float*)d_in[2];   // [4096,4096]
  const float* W_G      = (const float*)d_in[3];   // [256,256]
  const float* b_G      = (const float*)d_in[4];   // [256]

  float* out = (float*)d_out;
  float* out_cost = out;                                      // [4096,4096]
  float* out_ra   = out + (size_t)NROW * NROW;                // row_alignment
  float* out_ca   = out + 2ull * NROW * NROW;                 // column_alignment

  char* ws = (char*)d_ws;
  const size_t SZ_F = (size_t)NROW * NROW * sizeof(float);    // 64 MB
  const size_t SZ_B = (size_t)NROW * NROW * sizeof(__bf16);   // 32 MB
  const size_t SZ_V = (size_t)DDIM * NROW * sizeof(__bf16);   // 2 MB
  float*  costT   = (float*)(ws);                             // also holds colalignT f32
  __bf16* ra_bf   = (__bf16*)(ws + SZ_F);
  __bf16* cat_bf  = (__bf16*)(ws + SZ_F + SZ_B);
  __bf16* cvT     = (__bf16*)(ws + SZ_F + 2 * SZ_B);          // col_vecs^T bf16 [256,4096]
  __bf16* rvT     = (__bf16*)(ws + SZ_F + 2 * SZ_B + SZ_V);   // row_vecs^T bf16
  __bf16* WT      = (__bf16*)(ws + SZ_F + 2 * SZ_B + 2 * SZ_V);       // W_G^T bf16
  __bf16* beta    = (__bf16*)(ws + SZ_F + 2 * SZ_B + 2 * SZ_V + 131072);
  __bf16* alpha   = (__bf16*)((char*)beta + SZ_V);
  float*  acc     = (float*)((char*)alpha + SZ_V);            // [512]: v1acc | v2acc
  float*  yv      = acc + 512;

  dim3 tb(32, 8);

  zero_f32<<<1, 512, 0, stream>>>(acc, 512);

  // bf16 transposed operand staging
  transpose_to_bf16<<<dim3(DDIM / 32, NROW / 32), tb, 0, stream>>>(col_vecs, cvT, NROW, DDIM);
  transpose_to_bf16<<<dim3(DDIM / 32, NROW / 32), tb, 0, stream>>>(row_vecs, rvT, NROW, DDIM);
  transpose_to_bf16<<<dim3(DDIM / 32, DDIM / 32), tb, 0, stream>>>(W_G, WT, DDIM, DDIM);

  // costT for column-wise sparsemax
  transpose_f32<<<dim3(NROW / 32, NROW / 32), tb, 0, stream>>>(cost, costT, NROW, NROW);

  // row sparsemax -> out_ra (f32) + ra_bf;  column sparsemax (in place) -> costT + cat_bf
  sparsemax_rows<<<NROW, 256, 0, stream>>>(cost, out_ra, ra_bf);
  sparsemax_rows<<<NROW, 256, 0, stream>>>(costT, costT, cat_bf);

  // column_alignment[i,j] = colalignT[j,i]
  transpose_f32<<<dim3(NROW / 32, NROW / 32), tb, 0, stream>>>(costT, out_ca, NROW, NROW);

  // beta = row_align @ col_vecs ; alpha = colalignT @ row_vecs (WMMA + TDM staging)
  gemm_bf16_tdm<<<NROW / 16, 512, 0, stream>>>(ra_bf, cvT, beta, DDIM, NROW);
  gemm_bf16_tdm<<<NROW / 16, 512, 0, stream>>>(cat_bf, rvT, alpha, DDIM, NROW);

  // v1/v2 column sums of leakyrelu(X @ W_G + b_G)
  const int g2 = (NROW / 16) * (DDIM / 16) / 8;   // 512 blocks
  gemm_G_colsum<<<g2, 256, 0, stream>>>(beta, WT, b_G, acc);
  gemm_G_colsum<<<g2, 256, 0, stream>>>(alpha, WT, b_G, acc + 256);

  cosine_k<<<1, 256, 0, stream>>>(acc, yv);

  fill_y<<<2048, 256, 0, stream>>>((float4*)out_cost, yv, (NROW * NROW) / 4);
}
